// RippleGNNLayer_86981677679175
// MI455X (gfx1250) — compile-verified, matrix operand order checked by reading, not run on hardware
//
#include <hip/hip_runtime.h>
#include <hip/hip_bf16.h>
#include <math.h>

// Problem constants (from reference)
#define B_    4
#define N_    2048
#define DEVT  384
#define DIN   385      // D_DELTA + D_EVT
#define F_    128
#define FH    64       // F/2
#define H_    4
#define HF    512      // H*F
#define L_    2
#define KP1   416      // DIN padded to 13*32
#define NEG_BIG (-9000000000000000.0f)

typedef __attribute__((ext_vector_type(16))) __bf16 v16bf;
typedef __attribute__((ext_vector_type(8)))  float  v8f;
typedef __attribute__((ext_vector_type(4)))  int    v4i;

union BF16x16 { v16bf v; unsigned short s[16]; uint4 q[2]; };
union BF16x8  { unsigned short s[8]; uint4 q; };

// --- CDNA5 async global->LDS copy (probe-guarded) ---------------------------
#if defined(__has_builtin)
# if __has_builtin(__builtin_amdgcn_global_load_async_to_lds_b128) && \
     __has_builtin(__builtin_amdgcn_s_wait_asynccnt)
#  define HAS_ASYNC 1
# endif
#endif
#ifndef HAS_ASYNC
# define HAS_ASYNC 0
#endif

#if HAS_ASYNC
__device__ __forceinline__ void async_copy_b128(const void* g, void* l) {
  // builtin signature (from hipcc diagnostic): (v4i* src, v4i* lds_dst, imm, imm)
  __builtin_amdgcn_global_load_async_to_lds_b128((v4i*)g, (v4i*)l, 0, 0);
}
#endif

__device__ __forceinline__ unsigned short f2bf(float f) {
  unsigned int u = __float_as_uint(f);
  u += 0x7fffu + ((u >> 16) & 1u);            // round-to-nearest-even
  return (unsigned short)(u >> 16);
}
__device__ __forceinline__ float gelu_f(float x) {
  return 0.5f * x * (1.f + erff(x * 0.7071067811865476f));
}
__device__ __forceinline__ float sigmoid_f(float x) { return 1.f / (1.f + expf(-x)); }

// ---------------------------------------------------------------------------
// Fragment GEMM: one 16x16 output tile per wave (cols [wave*16, wave*16+16)).
//   A : bf16 row-major (16 rows, leading dim lda elements), LDS or global.
//   Wt: bf16 "fragment-major" packed weights: Wt[(kc*ncol + col)*32 + kk].
// Both fragments = two contiguous 16B runs -> b128 loads. No internal barriers.
// ---------------------------------------------------------------------------
__device__ __forceinline__ v8f gemm_frag(const unsigned short* A, int lda,
                                         const unsigned short* Wt, int ncol,
                                         int nChunks, int wave, int lane, v8f acc)
{
  const int m  = lane & 15;
  const int kb = (lane >> 4) << 3;
  const int c0 = wave << 4;
  if (c0 >= ncol) return acc;                        // wave-uniform
  const unsigned short* Ar = A + m * lda + kb;
  const unsigned short* Wc = Wt + ((size_t)(c0 + m)) * 32 + kb;
  for (int kc = 0; kc < nChunks; ++kc) {
    BF16x16 a, b;
    const unsigned short* ap = Ar + kc * 32;
    a.q[0] = *(const uint4*)(ap);
    a.q[1] = *(const uint4*)(ap + 16);
    const unsigned short* bp = Wc + (size_t)kc * ncol * 32;
    b.q[0] = *(const uint4*)(bp);
    b.q[1] = *(const uint4*)(bp + 16);
    acc = __builtin_amdgcn_wmma_f32_16x16x32_bf16(false, a.v, false, b.v,
                                                  (short)0, acc, false, false);
  }
  return acc;
}

// ---------------------------------------------------------------------------
// Pre-pack an f32 (K x ncol) weight into bf16 fragment-major layout
// ---------------------------------------------------------------------------
__global__ __launch_bounds__(256) void prep_kernel(const float* __restrict__ src,
                                                   unsigned short* __restrict__ dst,
                                                   int Ktot, int ncol, int nChunks)
{
  int total = nChunks * 32 * ncol;
  for (int i = blockIdx.x * 256 + threadIdx.x; i < total; i += gridDim.x * 256) {
    int kc  = i / (ncol * 32);
    int rem = i - kc * ncol * 32;
    int c   = rem >> 5;
    int kk  = rem & 31;
    int kg  = kc * 32 + kk;
    dst[i] = f2bf(kg < Ktot ? src[(size_t)kg * ncol + c] : 0.f);
  }
}

// ---------------------------------------------------------------------------
// Encoder: h = (gelu(gelu(x@mw1)@mw2)@mw3) * sigmoid(gelu(x@gw1)@gw2)
// ---------------------------------------------------------------------------
__global__ __launch_bounds__(256) void encode_kernel(
    const float* __restrict__ deltas, const float* __restrict__ evt,
    const unsigned short* __restrict__ mw1t, const float* __restrict__ mb1,
    const unsigned short* __restrict__ mw2t, const float* __restrict__ mb2,
    const unsigned short* __restrict__ mw3t, const float* __restrict__ mb3,
    const unsigned short* __restrict__ gw1t, const float* __restrict__ gb1,
    const unsigned short* __restrict__ gw2t, const float* __restrict__ gb2,
    unsigned short* __restrict__ h_bf)
{
  __shared__ unsigned short sA[16 * KP1];   // bf16 combined rows, K padded
  __shared__ unsigned short sX[16 * F_];    // bf16 intermediate activations
  __shared__ float          sM[16 * F_];    // f32 m3 storage
  const int tid = threadIdx.x, wave = tid >> 5, lane = tid & 31;
  const int row0 = blockIdx.x << 4;

  for (int i = tid; i < 16 * KP1; i += 256) {
    int r = i / KP1, k = i - r * KP1;
    int gr = row0 + r;
    float v = 0.f;
    if (k == 0)       v = deltas[gr];
    else if (k < DIN) v = evt[(size_t)gr * DEVT + (k - 1)];
    sA[i] = f2bf(v);
  }
  __syncthreads();
  const int rb  = (lane >> 4) << 3;
  const int col = (wave << 4) + (lane & 15);
  v8f z = {};

  v8f acc = gemm_frag(sA, KP1, mw1t, F_, 13, wave, lane, z);   // m1
  { float bs = mb1[col];
#pragma unroll
    for (int r = 0; r < 8; ++r) sX[(rb + r) * F_ + col] = f2bf(gelu_f(acc[r] + bs)); }
  __syncthreads();

  acc = gemm_frag(sX, F_, mw2t, F_, 4, wave, lane, z);         // m2
  __syncthreads();
  { float bs = mb2[col];
#pragma unroll
    for (int r = 0; r < 8; ++r) sX[(rb + r) * F_ + col] = f2bf(gelu_f(acc[r] + bs)); }
  __syncthreads();

  acc = gemm_frag(sX, F_, mw3t, F_, 4, wave, lane, z);         // m3 -> f32 LDS
  { float bs = mb3[col];
#pragma unroll
    for (int r = 0; r < 8; ++r) sM[(rb + r) * F_ + col] = acc[r] + bs; }

  acc = gemm_frag(sA, KP1, gw1t, FH, 13, wave, lane, z);       // g1
  __syncthreads();
  if (wave < 4) {
    float bs = gb1[col];
#pragma unroll
    for (int r = 0; r < 8; ++r) sX[(rb + r) * FH + col] = f2bf(gelu_f(acc[r] + bs));
  }
  __syncthreads();

  acc = gemm_frag(sX, FH, gw2t, F_, 2, wave, lane, z);         // g2; h = m3*g
  { float bs = gb2[col];
#pragma unroll
    for (int r = 0; r < 8; ++r) {
      float g = sigmoid_f(acc[r] + bs);
      h_bf[(size_t)(row0 + rb + r) * F_ + col] = f2bf(sM[(rb + r) * F_ + col] * g);
    } }
}

// ---------------------------------------------------------------------------
// Wh = h @ Wg[l,h] -> bf16 fragment-major; s = Wh.a_src, t = Wh.a_dst
// ---------------------------------------------------------------------------
__global__ __launch_bounds__(256) void wh_kernel(
    const unsigned short* __restrict__ h_bf,
    const unsigned short* __restrict__ Wgt_l,
    const float* __restrict__ asr, const float* __restrict__ adl,
    unsigned short* __restrict__ whb, float* __restrict__ s_g, float* __restrict__ t_g)
{
  __shared__ float sM[16 * F_];
  const int tid = threadIdx.x, wave = tid >> 5, lane = tid & 31;
  const int ntiles = N_ / 16;
  int bid = blockIdx.x;
  const int nt = bid % ntiles; bid /= ntiles;
  const int hh = bid % H_;     const int b = bid / H_;
  const int n0 = nt << 4;
  const int bh = b * H_ + hh;

  const unsigned short* Ab = h_bf + (size_t)(b * N_ + n0) * F_;
  v8f z = {};
  v8f acc = gemm_frag(Ab, F_, Wgt_l + (size_t)hh * 4 * F_ * 32, F_, 4, wave, lane, z);

  const int rb = (lane >> 4) << 3, col = (wave << 4) + (lane & 15);
  BF16x8 pk;
#pragma unroll
  for (int r = 0; r < 8; ++r) {
    float v = acc[r];
    sM[(rb + r) * F_ + col] = v;
    pk.s[r] = f2bf(v);
  }
  const int jc = n0 >> 5;
  unsigned short* dst = whb + (((size_t)bh * (N_ / 32) + jc) * F_ + col) * 32
                            + (n0 & 31) + rb;
  *(uint4*)dst = pk.q;                           // 8 contiguous bf16, 16B aligned

  __syncthreads();
  if (tid < 16) {
    const float* av = asr + hh * F_;
    const float* ad = adl + hh * F_;
    float ss = 0.f, tt = 0.f;
    for (int c = 0; c < F_; ++c) { float w = sM[tid * F_ + c]; ss += w * av[c]; tt += w * ad[c]; }
    s_g[(size_t)bh * N_ + n0 + tid] = ss;
    t_g[(size_t)bh * N_ + n0 + tid] = tt;
  }
}

// ---------------------------------------------------------------------------
// Flash-style masked GAT attention, j-chunks of 64 (2 WMMA per barrier phase).
// Adjacency (the only true HBM stream) is double-buffered into LDS with
// GLOBAL_LOAD_ASYNC_TO_LDS_B128 + s_wait_asynccnt when available.
// ---------------------------------------------------------------------------
#define CH 64
__global__ __launch_bounds__(256) void attn_kernel(
    const unsigned short* __restrict__ whb, const float* __restrict__ s_g,
    const float* __restrict__ t_g, const float* __restrict__ adj,
    unsigned short* __restrict__ hnext_bf)
{
  __shared__ float          sAdj[2][16 * CH];   // 8 KB double buffer
  __shared__ float          sS[16];
  __shared__ float          sT[CH];
  __shared__ float          sE[16 * CH];
  __shared__ unsigned short sP[16 * CH];
  __shared__ float sMax[16], sSum[16], sFac[16];
  const int tid = threadIdx.x, wave = tid >> 5, lane = tid & 31;
  const int ntiles = N_ / 16;
  int bid = blockIdx.x;
  const int nt = bid % ntiles; bid /= ntiles;
  const int hh = bid % H_;     const int b = bid / H_;
  const int i0 = nt << 4;
  const int bh = b * H_ + hh;
  const int NC = N_ / CH;

  if (tid < 16) {
    sS[tid]   = s_g[(size_t)bh * N_ + i0 + tid];
    sMax[tid] = -INFINITY;
    sSum[tid] = 0.f;
  }
  const float* adjb = adj + ((size_t)b * N_ + i0) * N_;   // 16 rows x N
  const int ar = tid >> 4;            // row 0..15
  const int ac = (tid & 15) << 2;     // 4-float column group

#if HAS_ASYNC
  async_copy_b128(adjb + (size_t)ar * N_ + ac, &sAdj[0][tid * 4]);   // chunk 0
#endif

  v8f acc = {};
  const int m = lane & 15, kb = (lane >> 4) << 3, c0 = wave << 4;
  const int rb = kb, col = c0 + m;
  const unsigned short* whl = whb + (size_t)bh * (N_ / 32) * F_ * 32;

  for (int jc = 0; jc < NC; ++jc) {
    const int j0 = jc * CH;
    const int buf = jc & 1;
    __syncthreads();                  // prev WMMA done with sP; buf^1 free
    if (tid < CH) sT[tid] = t_g[(size_t)bh * N_ + j0 + tid];
    __builtin_prefetch(whl + ((size_t)(jc + 1) * 2 * F_ + col) * 32 + kb, 0, 0);
#if HAS_ASYNC
    if (jc + 1 < NC) {                // issue next chunk, wait only for current
      async_copy_b128(adjb + (size_t)ar * N_ + (j0 + CH) + ac, &sAdj[buf ^ 1][tid * 4]);
      __builtin_amdgcn_s_wait_asynccnt(1);
    } else {
      __builtin_amdgcn_s_wait_asynccnt(0);
    }
#else
    for (int i = tid; i < 16 * CH; i += 256)
      sAdj[buf][i] = adjb[(size_t)(i >> 6) * N_ + j0 + (i & (CH - 1))];
#endif
    __syncthreads();                  // sT + adjacency chunk visible
    for (int i = tid; i < 16 * CH; i += 256) {   // masked leaky-relu logits
      int r = i >> 6, jj = i & (CH - 1);
      float e = sS[r] + sT[jj];
      e = e > 0.f ? e : 0.2f * e;
      sE[i] = (sAdj[buf][i] > 0.f) ? e : NEG_BIG;
    }
    __syncthreads();
    if (tid < 16) {                   // online softmax bookkeeping
      float mx = sMax[tid];
      for (int jj = 0; jj < CH; ++jj) mx = fmaxf(mx, sE[tid * CH + jj]);
      float fac = expf(sMax[tid] - mx);
      float sum = sSum[tid] * fac;
      for (int jj = 0; jj < CH; ++jj) {
        float p = expf(sE[tid * CH + jj] - mx);
        sP[tid * CH + jj] = f2bf(p);
        sum += p;
      }
      sMax[tid] = mx; sSum[tid] = sum; sFac[tid] = fac;
    }
    __syncthreads();
#pragma unroll
    for (int r = 0; r < 8; ++r) acc[r] *= sFac[rb + r];   // rescale accum
    acc = gemm_frag(sP, CH, whl + (size_t)jc * 2 * F_ * 32, F_, 2, wave, lane, acc);
  }
  __syncthreads();
#pragma unroll
  for (int r = 0; r < 8; ++r) {
    int row = rb + r;
    float v = acc[r] / sSum[row];
    v = v > 0.f ? v : expf(v) - 1.f;              // elu
    hnext_bf[(size_t)(b * N_ + i0 + row) * HF + hh * F_ + col] = f2bf(v);
  }
}

// ---------------------------------------------------------------------------
// Head-merge projection: h = hnext(B,N,512) @ pw[l] + pb[l]
// ---------------------------------------------------------------------------
__global__ __launch_bounds__(256) void proj_kernel(
    const unsigned short* __restrict__ hnext_bf,
    const unsigned short* __restrict__ pwt_l, const float* __restrict__ pbl,
    unsigned short* __restrict__ h_bf, float* __restrict__ h_f32)
{
  const int tid = threadIdx.x, wave = tid >> 5, lane = tid & 31;
  const int row0 = blockIdx.x << 4;
  const unsigned short* Ab = hnext_bf + (size_t)row0 * HF;
  v8f z = {};
  v8f acc = gemm_frag(Ab, HF, pwt_l, F_, 16, wave, lane, z);
  const int rb = (lane >> 4) << 3, col = (wave << 4) + (lane & 15);
  float bs = pbl[col];
#pragma unroll
  for (int r = 0; r < 8; ++r) {
    float v = acc[r] + bs;
    h_f32[(size_t)(row0 + rb + r) * F_ + col] = v;
    h_bf [(size_t)(row0 + rb + r) * F_ + col] = f2bf(v);
  }
}

// ---------------------------------------------------------------------------
// Output head: prop = gelu(h@ow1+ob1)@ow2+ob2; out = sigmoid(alpha)*deltas+beta*prop
// ---------------------------------------------------------------------------
__global__ __launch_bounds__(256) void out_kernel(
    const float* __restrict__ h_in, const float* __restrict__ ow1,
    const float* __restrict__ ob1, const float* __restrict__ ow2,
    const float* __restrict__ ob2, const float* __restrict__ alpha,
    const float* __restrict__ deltas, const float* __restrict__ beta,
    float* __restrict__ out)
{
  __shared__ float sRed[8][32];
  const int tid = threadIdx.x, wave = tid >> 5, lane = tid & 31;
  const int row = (blockIdx.x << 3) + wave;
  const float* hr = h_in + (size_t)row * F_;
  float part = 0.f;
#pragma unroll
  for (int t = 0; t < 2; ++t) {
    int j = lane + t * 32;
    float v = ob1[j];
    for (int k = 0; k < F_; ++k) v += hr[k] * ow1[k * FH + j];
    part += gelu_f(v) * ow2[j];
  }
  sRed[wave][lane] = part;
  __syncthreads();
  if (lane == 0) {
    float s = 0.f;
    for (int i = 0; i < 32; ++i) s += sRed[wave][i];
    s += ob2[0];
    int n = row & (N_ - 1);
    out[row] = sigmoid_f(alpha[n]) * deltas[row] + beta[0] * s;
  }
}

// ---------------------------------------------------------------------------
extern "C" void kernel_launch(void* const* d_in, const int* in_sizes, int n_in,
                              void* d_out, int out_size, void* d_ws, size_t ws_size,
                              hipStream_t stream)
{
  (void)in_sizes; (void)n_in; (void)out_size; (void)ws_size;
  const float* deltas = (const float*)d_in[0];
  const float* evt    = (const float*)d_in[1];
  const float* adj    = (const float*)d_in[2];
  const float* mw1 = (const float*)d_in[3];  const float* mb1 = (const float*)d_in[4];
  const float* mw2 = (const float*)d_in[5];  const float* mb2 = (const float*)d_in[6];
  const float* mw3 = (const float*)d_in[7];  const float* mb3 = (const float*)d_in[8];
  const float* gw1 = (const float*)d_in[9];  const float* gb1 = (const float*)d_in[10];
  const float* gw2 = (const float*)d_in[11]; const float* gb2 = (const float*)d_in[12];
  const float* Wg    = (const float*)d_in[13];
  const float* a_src = (const float*)d_in[14];
  const float* a_dst = (const float*)d_in[15];
  const float* pw  = (const float*)d_in[16]; const float* pb  = (const float*)d_in[17];
  const float* ow1 = (const float*)d_in[18]; const float* ob1 = (const float*)d_in[19];
  const float* ow2 = (const float*)d_in[20]; const float* ob2 = (const float*)d_in[21];
  const float* alpha = (const float*)d_in[22];
  const float* beta  = (const float*)d_in[23];

  char* ws = (char*)d_ws;
  size_t off = 0;
  auto carve = [&](size_t bytes) -> void* {
    void* p = ws + off;
    off = (off + bytes + 255) & ~(size_t)255;
    return p;
  };
  unsigned short* h_bf     = (unsigned short*)carve((size_t)B_ * N_ * F_ * 2);
  float*          h_f32    = (float*)carve((size_t)B_ * N_ * F_ * 4);
  unsigned short* whb      = (unsigned short*)carve((size_t)B_ * H_ * N_ * F_ * 2);
  unsigned short* hnext_bf = (unsigned short*)carve((size_t)B_ * N_ * HF * 2);
  float*          s_g      = (float*)carve((size_t)B_ * H_ * N_ * 4);
  float*          t_g      = (float*)carve((size_t)B_ * H_ * N_ * 4);
  unsigned short* mw1t = (unsigned short*)carve((size_t)13 * F_ * 32 * 2);
  unsigned short* mw2t = (unsigned short*)carve((size_t)4  * F_ * 32 * 2);
  unsigned short* mw3t = (unsigned short*)carve((size_t)4  * F_ * 32 * 2);
  unsigned short* gw1t = (unsigned short*)carve((size_t)13 * FH * 32 * 2);
  unsigned short* gw2t = (unsigned short*)carve((size_t)2  * F_ * 32 * 2);
  unsigned short* Wgt  = (unsigned short*)carve((size_t)L_ * H_ * 4 * F_ * 32 * 2);
  unsigned short* pwt  = (unsigned short*)carve((size_t)L_ * 16 * F_ * 32 * 2);

  auto prep = [&](const float* src, unsigned short* dst, int Ktot, int ncol, int nch) {
    int total = nch * 32 * ncol;
    prep_kernel<<<(total + 255) / 256, 256, 0, stream>>>(src, dst, Ktot, ncol, nch);
  };
  prep(mw1, mw1t, DIN, F_, 13);
  prep(mw2, mw2t, F_,  F_, 4);
  prep(mw3, mw3t, F_,  F_, 4);
  prep(gw1, gw1t, DIN, FH, 13);
  prep(gw2, gw2t, FH,  F_, 2);
  for (int l = 0; l < L_; ++l) {
    for (int hh = 0; hh < H_; ++hh)
      prep(Wg + ((size_t)(l * H_ + hh)) * F_ * F_,
           Wgt + ((size_t)(l * H_ + hh)) * 4 * F_ * 32, F_, F_, 4);
    prep(pw + (size_t)l * HF * F_, pwt + (size_t)l * 16 * F_ * 32, HF, F_, 16);
  }

  encode_kernel<<<(B_ * N_) / 16, 256, 0, stream>>>(
      deltas, evt, mw1t, mb1, mw2t, mb2, mw3t, mb3, gw1t, gb1, gw2t, gb2, h_bf);

  for (int l = 0; l < L_; ++l) {
    wh_kernel<<<B_ * H_ * (N_ / 16), 256, 0, stream>>>(
        h_bf, Wgt + (size_t)l * H_ * 4 * F_ * 32,
        a_src + (size_t)l * H_ * F_, a_dst + (size_t)l * H_ * F_,
        whb, s_g, t_g);
    attn_kernel<<<B_ * H_ * (N_ / 16), 256, 0, stream>>>(whb, s_g, t_g, adj, hnext_bf);
    proj_kernel<<<(B_ * N_) / 16, 256, 0, stream>>>(
        hnext_bf, pwt + (size_t)l * 16 * F_ * 32, pb + (size_t)l * F_, h_bf, h_f32);
  }

  out_kernel<<<(B_ * N_) / 8, 256, 0, stream>>>(
      h_f32, ow1, ob1, ow2, ob2, alpha, deltas, beta, (float*)d_out);
}
